// AWDLSTM_1846835937951
// MI455X (gfx1250) — matrix operand order repaired and checked by reading, not compiled, not run on hardware
//
#include <hip/hip_runtime.h>
#include <hip/hip_bf16.h>
#include <hip/hip_fp16.h>

// ---------------------------------------------------------------------------
// AWD-LSTM for MI455X (gfx1250, wave32, WMMA + TDM).
//  B=32, S=1024, E=H=512, gates 4H=2048, M=B*S=32768.
//
//  Phase 1: pre = X @ W_ih^T + b_ih + b_hh     (WMMA f16 GEMM, HBM-bound)
//  Phase 2: serial recurrence, 16 WGs per direction, w_hh slice resident in
//           LDS as f16 (320KB LDS/WGP). h_t kept in global as f16 and
//           broadcast into each WGP's LDS via the Tensor Data Mover
//           (tensor_load_to_lds + s_wait_tensorcnt), with TDM row padding.
//           Device-scope phase barrier per timestep.
//  Phase 3: fc head + mish (tiny, VALU).
// ---------------------------------------------------------------------------

typedef __attribute__((ext_vector_type(16))) _Float16 v16h;
typedef __attribute__((ext_vector_type(8)))  float    v8f;
typedef __attribute__((ext_vector_type(4)))  unsigned int v4u;
typedef __attribute__((ext_vector_type(8)))  int      v8i;
typedef __attribute__((ext_vector_type(4)))  int      v4i;

#define LSTM_B   32
#define LSTM_S   1024
#define LSTM_H   512
#define LSTM_G   2048            // 4H
#define LSTM_M   (LSTM_B * LSTM_S)

// ---- CDNA5 WMMA fragment helpers (contiguous k order, same in A and B)

// A-matrix 16x32 f16: lane holds row M=lane&15.
// lane<16 : halves 0..15 -> K=k0+0..15
// lane>=16: halves 0..15 -> K=k0+16..31
__device__ inline v16h load_frag_A(const _Float16* lds, int m0, int k0,
                                   int ldk, int lane) {
  int m  = m0 + (lane & 15);
  int kb = k0 + ((lane >> 4) << 4);
  const _Float16* p = lds + m * ldk + kb;
  v16h a;
#pragma unroll
  for (int t = 0; t < 16; ++t) a[t] = p[t];
  return a;
}

// B-matrix 32x16 f16 (B = W^T, stored as rows of W: [n][k]):
// lane holds column N=lane&15; lanes 0-15: K=k0..k0+15, lanes 16-31: +16.
__device__ inline v16h load_frag_B(const _Float16* lds, int n0, int k0,
                                   int ldk, int lane) {
  int n  = n0 + (lane & 15);
  int kb = k0 + ((lane >> 4) << 4);
  const _Float16* p = lds + n * ldk + kb;
  v16h b;
#pragma unroll
  for (int t = 0; t < 16; ++t) b[t] = p[t];
  return b;
}

__device__ inline v8f wmma_f16(v16h a, v16h b, v8f c) {
  return __builtin_amdgcn_wmma_f32_16x16x32_f16(false, a, false, b,
                                                (short)0, c, false, false);
}

__device__ inline float sig_f(float x)  { return 1.0f / (1.0f + __expf(-x)); }
__device__ inline float tanh_f(float x) {
  float e = __expf(2.0f * x);
  return 1.0f - 2.0f / (e + 1.0f);
}

// ---- device-scope phase barrier (cnt returns to 0, phase monotonic) --------
__device__ inline void grid_barrier(int* cnt, int* ph, int nblk) {
  __syncthreads();
  if (threadIdx.x == 0) {
    int p = __hip_atomic_load(ph, __ATOMIC_ACQUIRE, __HIP_MEMORY_SCOPE_AGENT);
    int a = __hip_atomic_fetch_add(cnt, 1, __ATOMIC_ACQ_REL,
                                   __HIP_MEMORY_SCOPE_AGENT);
    if (a == nblk - 1) {
      __hip_atomic_store(cnt, 0, __ATOMIC_RELAXED, __HIP_MEMORY_SCOPE_AGENT);
      __hip_atomic_fetch_add(ph, 1, __ATOMIC_ACQ_REL,
                             __HIP_MEMORY_SCOPE_AGENT);
    } else {
      while (__hip_atomic_load(ph, __ATOMIC_ACQUIRE,
                               __HIP_MEMORY_SCOPE_AGENT) == p) {
        __builtin_amdgcn_s_sleep(2);
      }
    }
  }
  __syncthreads();
}

// ===========================================================================
// Phase 1: pre[m,n] = sum_k X[m,k]*W[n,k] + b1[n] + b2[n]
//   block 256 (8 waves), tile 64x128, BK=32, f32->f16 staged via LDS.
// ===========================================================================
#define ALD 40   // padded LDS row (halfs) to dodge bank conflicts
__global__ __launch_bounds__(256) void gemm_pre_kernel(
    const float* __restrict__ X, const float* __restrict__ W,
    const float* __restrict__ b1, const float* __restrict__ b2,
    float* __restrict__ C, int K) {
  __shared__ _Float16 As[64 * ALD];
  __shared__ _Float16 Bs[128 * ALD];
  const int tid  = threadIdx.x;
  const int wave = tid >> 5, lane = tid & 31;
  const int m0 = blockIdx.x * 64;
  const int n0 = blockIdx.y * 128;

  v8f acc[4] = {};
  for (int k0 = 0; k0 < K; k0 += 32) {
    // stage A 64x32 (2 float4 / thread)
#pragma unroll
    for (int it = 0; it < 2; ++it) {
      int i4 = tid + it * 256;                 // 0..511
      int row = i4 >> 3, col = (i4 & 7) << 2;
      float4 v = *(const float4*)(X + (long)(m0 + row) * K + k0 + col);
      _Float16* d = As + row * ALD + col;
      d[0] = (_Float16)v.x; d[1] = (_Float16)v.y;
      d[2] = (_Float16)v.z; d[3] = (_Float16)v.w;
    }
    // stage B 128x32 (4 float4 / thread)
#pragma unroll
    for (int it = 0; it < 4; ++it) {
      int i4 = tid + it * 256;                 // 0..1023
      int row = i4 >> 3, col = (i4 & 7) << 2;
      float4 v = *(const float4*)(W + (long)(n0 + row) * K + k0 + col);
      _Float16* d = Bs + row * ALD + col;
      d[0] = (_Float16)v.x; d[1] = (_Float16)v.y;
      d[2] = (_Float16)v.z; d[3] = (_Float16)v.w;
    }
    if (k0 + 32 < K) {    // hint next k-block into cache (global_prefetch_b8)
      __builtin_prefetch(X + (long)(m0 + (tid >> 2)) * K + k0 + 32, 0, 1);
      __builtin_prefetch(W + (long)(n0 + (tid >> 1)) * K + k0 + 32, 0, 1);
    }
    __syncthreads();
    v16h bf = load_frag_B(Bs, wave * 16, 0, ALD, lane);
#pragma unroll
    for (int mt = 0; mt < 4; ++mt) {
      v16h af = load_frag_A(As, mt * 16, 0, ALD, lane);
      acc[mt] = wmma_f16(af, bf, acc[mt]);
    }
    __syncthreads();
  }
  // epilogue: C/D layout -> n = lane&15, m = r + 8*(lane>=16)
  const int n    = n0 + wave * 16 + (lane & 15);
  const float bs = b1[n] + b2[n];
  const int mb   = m0 + ((lane >> 4) << 3);
#pragma unroll
  for (int mt = 0; mt < 4; ++mt)
#pragma unroll
    for (int r = 0; r < 8; ++r)
      C[(long)(mb + mt * 16 + r) * LSTM_G + n] = acc[mt][r] + bs;
}

// ===========================================================================
// Phase 2: recurrence. gridDim=(16,2): x = hidden slice (32 units), y = dir.
//  LDS: w_hh slice 128x512 f16 (padded WLD), h_t 32x512 f16 (padded, filled
//  by the TDM), gates 32x128 f32, c 32x32 f32.
//  Global h buffers are f16 ping-pong; one device barrier per step.
// ===========================================================================
#define WLD 520
#define REC_SMEM_W  (128 * WLD * 2)                 // 133120
#define REC_SMEM_H  (32 * WLD * 2)                  //  33280
#define REC_SMEM_G  (32 * 128 * 4)                  //  16384
#define REC_SMEM_C  (32 * 32 * 4)                   //   4096
#define REC_SMEM    (REC_SMEM_W + REC_SMEM_H + REC_SMEM_G + REC_SMEM_C)
#define HBUF_LDS_OFF REC_SMEM_W                     // byte offset of hbuf

// Issue a TDM copy of the 32x512 f16 h-matrix (row-major, stride 512) from
// global into LDS at HBUF_LDS_OFF, padding each 1024B row with 16B so rows
// land WLD halfs apart.  D# layout per CDNA5 ISA.
// This toolchain exposes the 6-arg builtin:
//   (uint32x4 g0, int32x8 g1, int32x4 g2, int32x4 g3, int32x8 g4, i32 cpol)
__device__ inline void tdm_load_h(const _Float16* hr) {
  unsigned long long ga = (unsigned long long)(uintptr_t)hr;
  v4u g0;
  g0[0] = 1u;                                  // count=1 (valid user D#)
  g0[1] = (unsigned)HBUF_LDS_OFF;              // lds_addr (bytes)
  g0[2] = (unsigned)(ga & 0xffffffffu);        // global_addr[31:0]
  g0[3] = (unsigned)((ga >> 32) & 0x1ffffffu)  // global_addr[56:32]
        | (2u << 30);                          // type = 2 ("image")
  v8i g1;
  g1[0] = (1 << 16)                            // data_size = 2 bytes
        | (1 << 20)                            // pad_enable
        | (7 << 22)                            // pad_interval: 256 DW = 1 row
        | (3 << 25);                           // pad_amount: 4 DW = 16 B
  g1[1] = (LSTM_H & 0xffff) << 16;             // tensor_dim0 = 512 (lo)
  g1[2] = (LSTM_B << 16);                      // dim0 hi | tensor_dim1 = 32
  g1[3] = (LSTM_H << 16);                      // dim1 hi | tile_dim0 = 512
  g1[4] = LSTM_B;                              // tile_dim1 = 32, tile_dim2=0
  g1[5] = LSTM_H;                              // tensor_dim0_stride = 512
  g1[6] = 0;
  g1[7] = 0;
  v4i gz4 = {0, 0, 0, 0};                      // 2D tensor: groups 2/3 unused
  v8i gz8 = {0, 0, 0, 0, 0, 0, 0, 0};
  __builtin_amdgcn_tensor_load_to_lds(g0, g1, gz4, gz4, gz8, 0);
  __builtin_amdgcn_s_wait_tensorcnt(0);
}

__global__ __launch_bounds__(256) void lstm_rec_kernel(
    const float* __restrict__ pre_f, const float* __restrict__ pre_b,
    const float* __restrict__ whh_f, const float* __restrict__ whh_b,
    const float* __restrict__ mask_f,          // DropConnect (dir 0 only)
    _Float16* hglob_f, _Float16* hglob_b,      // 2*32*512 halfs each
    float* seq_out, int seq_ld, int col_f, int col_b,
    int S, int* bar) {
  extern __shared__ __align__(16) char smem[];
  _Float16* wbuf = (_Float16*)smem;
  _Float16* hbuf = (_Float16*)(smem + REC_SMEM_W);
  float*    gbuf = (float*)(smem + REC_SMEM_W + REC_SMEM_H);
  float*    cbuf = (float*)(smem + REC_SMEM_W + REC_SMEM_H + REC_SMEM_G);

  const int tid  = threadIdx.x;
  const int dir  = blockIdx.y;
  const int hid0 = blockIdx.x * 32;
  const float* pre  = dir ? pre_b  : pre_f;
  const float* whh  = dir ? whh_b  : whh_f;
  const float* mask = dir ? nullptr : mask_f;
  _Float16*    hg   = dir ? hglob_b : hglob_f;
  const int colbase = dir ? col_b : col_f;
  int* cnt = bar + dir * 2;
  int* ph  = bar + dir * 2 + 1;
  const int nblk = gridDim.x;

  // --- stage w_hh slice into LDS as f16 (once); local col c:
  //     gate = c>>5, global gate column = gate*512 + hid0 + (c&31)
  for (int i = tid; i < 128 * 512 / 4; i += 256) {
    int idx = i * 4;
    int c = idx >> 9, k = idx & 511;
    int ng = (c >> 5) * LSTM_H + hid0 + (c & 31);
    float4 w4 = *(const float4*)(whh + (long)ng * LSTM_H + k);
    if (mask) {
      float4 m4 = *(const float4*)(mask + (long)ng * LSTM_H + k);
      w4.x *= m4.x; w4.y *= m4.y; w4.z *= m4.z; w4.w *= m4.w;
    }
    _Float16* d = wbuf + c * WLD + k;
    d[0] = (_Float16)w4.x; d[1] = (_Float16)w4.y;
    d[2] = (_Float16)w4.z; d[3] = (_Float16)w4.w;
  }
  // zero c state and our slice of h buffer 0
  for (int i = tid; i < 32 * 32; i += 256) cbuf[i] = 0.0f;
  for (int i = tid; i < 32 * 32; i += 256)
    hg[(i >> 5) * LSTM_H + hid0 + (i & 31)] = (_Float16)0.0f;
  __syncthreads();
  grid_barrier(cnt, ph, nblk);

  const int wave = tid >> 5, lane = tid & 31;
  const int n0 = wave * 16;                      // local gate column tile

  for (int step = 0; step < S; ++step) {
    const int t = dir ? (S - 1 - step) : step;
    const _Float16* hr = hg + (step & 1) * (LSTM_B * LSTM_H);
    _Float16*       hw = hg + ((step + 1) & 1) * (LSTM_B * LSTM_H);

    // DMA h_t into LDS via the Tensor Data Mover (wave 0 only; TENSORcnt)
    if (tid < 32) tdm_load_h(hr);
    __syncthreads();

    // h @ w_hh^T for our 128 columns: 2 m-tiles x 8 n-tiles, wave -> n-tile
    v8f acc0 = {}, acc1 = {};
    for (int k0 = 0; k0 < LSTM_H; k0 += 32) {
      v16h bf = load_frag_B(wbuf, n0, k0, WLD, lane);
      v16h a0 = load_frag_A(hbuf,  0, k0, WLD, lane);
      v16h a1 = load_frag_A(hbuf, 16, k0, WLD, lane);
      acc0 = wmma_f16(a0, bf, acc0);
      acc1 = wmma_f16(a1, bf, acc1);
    }
    // add precomputed input contribution, park gates in LDS
    {
      int nl = n0 + (lane & 15);
      int ng = (nl >> 5) * LSTM_H + hid0 + (nl & 31);
      int mb = (lane >> 4) << 3;
      const float* prow = pre + (long)t * LSTM_G + ng;
#pragma unroll
      for (int r = 0; r < 8; ++r) {
        int m = mb + r;
        gbuf[m * 128 + nl] =
            acc0[r] + prow[(long)m * (long)S * LSTM_G];
        gbuf[(m + 16) * 128 + nl] =
            acc1[r] + prow[(long)(m + 16) * (long)S * LSTM_G];
      }
    }
    __syncthreads();

    // elementwise cell update for our 32 hidden units (i,f,g,o order)
    for (int i = tid; i < 32 * 32; i += 256) {
      int m = i >> 5, j = i & 31;
      float gi = gbuf[m * 128 + j];
      float gf = gbuf[m * 128 + 32 + j];
      float gg = gbuf[m * 128 + 64 + j];
      float go = gbuf[m * 128 + 96 + j];
      float c  = cbuf[i];
      c = sig_f(gf) * c + sig_f(gi) * tanh_f(gg);
      cbuf[i] = c;
      float h = sig_f(go) * tanh_f(c);
      hw[m * LSTM_H + hid0 + j] = (_Float16)h;
      if (seq_out)
        seq_out[((long)m * S + t) * seq_ld + colbase + j] = h;
    }
    grid_barrier(cnt, ph, nblk);   // publish h_{t+1} to the other 15 WGPs
  }
}

// ===========================================================================
// Phase 3: out = mish( (0.5*(hf+hb)) @ fc_w^T + fc_b )   [32 x 512]
// ===========================================================================
__global__ __launch_bounds__(256) void head_kernel(
    const _Float16* __restrict__ hf, const _Float16* __restrict__ hb,
    const float* __restrict__ fcw, const float* __restrict__ fcb,
    float* __restrict__ out) {
  int idx = blockIdx.x * blockDim.x + threadIdx.x;   // 0..16383
  int m = idx >> 9, n = idx & 511;
  const float* wr  = fcw + (long)n * LSTM_H;
  const _Float16* hfr = hf + m * LSTM_H;
  const _Float16* hbr = hb + m * LSTM_H;
  float s = fcb[n];
  for (int k = 0; k < LSTM_H; ++k)
    s += 0.5f * ((float)hfr[k] + (float)hbr[k]) * wr[k];
  float sp = (s > 20.0f) ? s : log1pf(__expf(s));
  out[idx] = s * tanhf(sp);
}

__global__ void init_bar_kernel(int* bar) {
  if (threadIdx.x < 64) bar[threadIdx.x] = 0;
}

// ===========================================================================
extern "C" void kernel_launch(void* const* d_in, const int* in_sizes, int n_in,
                              void* d_out, int out_size, void* d_ws,
                              size_t ws_size, hipStream_t stream) {
  (void)in_sizes; (void)n_in; (void)out_size; (void)ws_size;
  const float* x       = (const float*)d_in[0];
  const float* w_ih_f0 = (const float*)d_in[1];
  const float* w_hh_f0 = (const float*)d_in[2];
  const float* b_ih_f0 = (const float*)d_in[3];
  const float* b_hh_f0 = (const float*)d_in[4];
  const float* w_ih_b0 = (const float*)d_in[5];
  const float* w_hh_b0 = (const float*)d_in[6];
  const float* b_ih_b0 = (const float*)d_in[7];
  const float* b_hh_b0 = (const float*)d_in[8];
  const float* w_ih_f1 = (const float*)d_in[9];
  const float* w_hh_f1 = (const float*)d_in[10];
  const float* b_ih_f1 = (const float*)d_in[11];
  const float* b_hh_f1 = (const float*)d_in[12];
  const float* w_ih_b1 = (const float*)d_in[13];
  const float* w_hh_b1 = (const float*)d_in[14];
  const float* b_ih_b1 = (const float*)d_in[15];
  const float* b_hh_b1 = (const float*)d_in[16];
  const float* fc_w    = (const float*)d_in[17];
  const float* fc_b    = (const float*)d_in[18];
  const float* mask    = (const float*)d_in[19];

  char* ws = (char*)d_ws;
  const size_t PRE_BYTES = (size_t)LSTM_M * LSTM_G * sizeof(float);   // 256 MB
  const size_t X1_BYTES  = (size_t)LSTM_M * 1024 * sizeof(float);     // 128 MB
  float*    pre_f = (float*)(ws);
  float*    pre_b = (float*)(ws + PRE_BYTES);
  float*    x1    = (float*)(ws + 2 * PRE_BYTES);
  _Float16* h0f   = (_Float16*)(ws + 2 * PRE_BYTES + X1_BYTES);
  _Float16* h0b   = h0f + 2 * LSTM_B * LSTM_H;
  _Float16* h1f   = h0b + 2 * LSTM_B * LSTM_H;
  _Float16* h1b   = h1f + 2 * LSTM_B * LSTM_H;
  int*      bar   = (int*)(h1b + 2 * LSTM_B * LSTM_H);

  init_bar_kernel<<<1, 64, 0, stream>>>(bar);

  dim3 gg(LSTM_M / 64, LSTM_G / 128);   // (512, 16)
  // layer 0 input projections
  gemm_pre_kernel<<<gg, 256, 0, stream>>>(x, w_ih_f0, b_ih_f0, b_hh_f0, pre_f, 512);
  gemm_pre_kernel<<<gg, 256, 0, stream>>>(x, w_ih_b0, b_ih_b0, b_hh_b0, pre_b, 512);
  // layer 0 recurrence (both directions concurrently), writes x1
  lstm_rec_kernel<<<dim3(16, 2), 256, REC_SMEM, stream>>>(
      pre_f, pre_b, w_hh_f0, w_hh_b0, mask, h0f, h0b,
      x1, 1024, 0, 512, LSTM_S, bar);
  // layer 1 input projections (K = 1024), reuse pre buffers
  gemm_pre_kernel<<<gg, 256, 0, stream>>>(x1, w_ih_f1, b_ih_f1, b_hh_f1, pre_f, 1024);
  gemm_pre_kernel<<<gg, 256, 0, stream>>>(x1, w_ih_b1, b_ih_b1, b_hh_b1, pre_b, 1024);
  // layer 1 recurrence; final h lands in buffer (S&1)==0 (base pointer)
  lstm_rec_kernel<<<dim3(16, 2), 256, REC_SMEM, stream>>>(
      pre_f, pre_b, w_hh_f1, w_hh_b1, nullptr, h1f, h1b,
      nullptr, 0, 0, 0, LSTM_S, bar);
  // head
  head_kernel<<<LSTM_B * LSTM_H / 256, 256, 0, stream>>>(
      h1f, h1b, fc_w, fc_b, (float*)d_out);
}
